// GNNPolicy_33629593927742
// MI455X (gfx1250) — matrix operand order repaired and checked by reading, not compiled, not run on hardware
//
#include <hip/hip_runtime.h>

// ---------------------------------------------------------------------------
// MI455X (gfx1250) GNN forward.
//  - All 64x64 dense matmuls use v_wmma_f32_16x16x32_f16 (fp16 A/B, f32 acc).
//  - Per-edge GEMMs algebraically hoisted to node level.
//  - Edge stage: gather + LayerNorm(64) via wave32 shuffles + f32 L2 atomics.
//  - GEMM epilogue transposed through LDS for b128-coalesced global stores.
// ---------------------------------------------------------------------------

typedef __attribute__((ext_vector_type(16))) _Float16 v16h;
typedef __attribute__((ext_vector_type(8)))  _Float16 v8h;
typedef __attribute__((ext_vector_type(4)))  _Float16 v4h;
typedef __attribute__((ext_vector_type(8)))  float    v8f;
typedef __attribute__((ext_vector_type(4)))  float    v4f;

struct HPair { v8h lo, hi; };

// ---------------------------------------------------------------------------
// Y[M,64] = act( A[M,64](f16) @ W[64,64](f32, K-major) [+ Cin] [+ bias] )
// One wave -> 16 rows x 64 cols (4 N-tiles x 2 K-steps = 8 WMMA).
// Block = 256 threads = 8 waves = 128 rows.
// W staged in LDS pre-swizzled into B-fragment order (2x ds_load_b128/frag).
// Accumulators staged through LDS so all global I/O is b128 vectors.
// ---------------------------------------------------------------------------
template <int HAS_CIN, int HAS_BIAS, int RELU, int OUT_F, int OUT_H>
__global__ __launch_bounds__(256) void gemm64(
    const _Float16* __restrict__ A, const float* __restrict__ W,
    const float* __restrict__ bias, const float* __restrict__ Cin,
    float* __restrict__ Yf, _Float16* __restrict__ Yh, int M)
{
  __shared__ _Float16 sW[8 * 512];       // 8 frags x 32 lanes x 16 halves (8 KB)
  __shared__ float    sOut[8 * 16 * 64]; // per-wave 16x64 f32 tile (32 KB)
  const int tid = threadIdx.x;

  // Fill B fragments in swizzled order: frag f = nt*2+kh,
  // element t of lane L holds W[kh*32 + (L>>4)*16 + t][nt*16 + (L&15)].
  for (int idx = tid; idx < 4096; idx += 256) {
    const int t   = idx & 15;
    const int ln  = (idx >> 4) & 31;
    const int f   = idx >> 9;
    const int nt  = f >> 1, kh = f & 1;
    const int k   = kh * 32 + (ln >> 4) * 16 + t;
    const int n   = nt * 16 + (ln & 15);
    sW[idx] = (_Float16)W[k * 64 + n];
  }
  __syncthreads();

  const int wave = tid >> 5, lane = tid & 31;
  const int half = lane >> 4, l16 = lane & 15;
  const int row0 = ((int)blockIdx.x * 8 + wave) * 16;
  if (row0 >= M) return;

  // A fragments: row = row0 + l16; per K-step two contiguous 16B loads:
  //   t=0..7  <-> K = kh*32 + half*16 + t
  //   t=8..15 <-> K = kh*32 + half*16 + t
  const int arow = row0 + l16;
  const _Float16* ap = A + (size_t)arow * 64;
  if (arow + 2048 + 15 < M) __builtin_prefetch(A + (size_t)(arow + 2048) * 64, 0, 1);
  v16h Af[2];
#pragma unroll
  for (int kh = 0; kh < 2; ++kh) {
    HPair hp; hp.lo = v8h{}; hp.hi = v8h{};
    if (arow < M) {
      hp.lo = *(const v8h*)(ap + kh * 32 + half * 16);
      hp.hi = *(const v8h*)(ap + kh * 32 + half * 16 + 8);
    }
    Af[kh] = __builtin_bit_cast(v16h, hp);
  }

  float* so = sOut + wave * (16 * 64);
#pragma unroll
  for (int nt = 0; nt < 4; ++nt) {
    const v16h B0 = *(const v16h*)(sW + (nt * 2 + 0) * 512 + lane * 16);
    const v16h B1 = *(const v16h*)(sW + (nt * 2 + 1) * 512 + lane * 16);
    v8f acc = {};
    acc = __builtin_amdgcn_wmma_f32_16x16x32_f16(false, Af[0], false, B0,
                                                 (short)0, acc, false, false);
    acc = __builtin_amdgcn_wmma_f32_16x16x32_f16(false, Af[1], false, B1,
                                                 (short)0, acc, false, false);
    // D layout: lane col = nt*16+l16, rows = half*8 + r
#pragma unroll
    for (int r = 0; r < 8; ++r)
      so[(half * 8 + r) * 64 + nt * 16 + l16] = acc[r];
  }
  asm volatile("s_wait_dscnt 0x0" ::: "memory");  // same-wave LDS RAW

  // Coalesced epilogue: lane handles half a row (32 contiguous floats).
  const int rr = lane >> 1;
  const int ch = (lane & 1) * 32;
  const int m  = row0 + rr;
  if (m < M) {
    const float* srow = so + rr * 64 + ch;
    float v[32];
#pragma unroll
    for (int c = 0; c < 32; c += 4) {
      const v4f t = *(const v4f*)(srow + c);
      v[c] = t[0]; v[c+1] = t[1]; v[c+2] = t[2]; v[c+3] = t[3];
    }
    if (HAS_CIN) {
#pragma unroll
      for (int c = 0; c < 32; c += 4) {
        const v4f t = *(const v4f*)(Cin + (size_t)m * 64 + ch + c);
        v[c] += t[0]; v[c+1] += t[1]; v[c+2] += t[2]; v[c+3] += t[3];
      }
    }
    if (HAS_BIAS) {
#pragma unroll
      for (int c = 0; c < 32; c += 4) {
        const v4f t = *(const v4f*)(bias + ch + c);
        v[c] += t[0]; v[c+1] += t[1]; v[c+2] += t[2]; v[c+3] += t[3];
      }
    }
    if (RELU) {
#pragma unroll
      for (int c = 0; c < 32; ++c) v[c] = fmaxf(v[c], 0.f);
    }
    if (OUT_F) {
#pragma unroll
      for (int c = 0; c < 32; c += 4) {
        v4f o; o[0] = v[c]; o[1] = v[c+1]; o[2] = v[c+2]; o[3] = v[c+3];
        *(v4f*)(Yf + (size_t)m * 64 + ch + c) = o;
      }
    }
    if (OUT_H) {
#pragma unroll
      for (int c = 0; c < 32; c += 8) {
        v8h o;
#pragma unroll
        for (int t = 0; t < 8; ++t) o[t] = (_Float16)v[c + t];
        *(v8h*)(Yh + (size_t)m * 64 + ch + c) = o;
      }
    }
  }
}

// ---------------------------------------------------------------------------
// Per-edge: m = Pr[i] + Pl[j] + eb*We ; relu(LN(m)) -> atomic add into agg[i].
// One wave per edge (2 features per lane), LN via shuffle tree.
// Pr/Pl/agg all fit in the 192MB L2 -> gathers & fadd atomics resolve at L2.
// ---------------------------------------------------------------------------
__global__ __launch_bounds__(256) void edge_msg(
    const int* __restrict__ tgt, const int* __restrict__ src,
    const float* __restrict__ Pr, const float* __restrict__ Pl,
    const float* __restrict__ We, const float* __restrict__ ebp,
    const float* __restrict__ fg, const float* __restrict__ fb,
    float* __restrict__ agg, int E)
{
  const int e = (int)blockIdx.x * 8 + (int)(threadIdx.x >> 5);
  if (e >= E) return;
  const int lane = threadIdx.x & 31;
  const int i = tgt[e], j = src[e];
  const float eb = ebp[0];
  const int n0 = lane, n1 = lane + 32;
  const size_t ri = (size_t)i * 64, rj = (size_t)j * 64;
  float m0 = Pr[ri + n0] + Pl[rj + n0] + eb * We[n0];
  float m1 = Pr[ri + n1] + Pl[rj + n1] + eb * We[n1];
  float s = m0 + m1;
#pragma unroll
  for (int o = 16; o >= 1; o >>= 1) s += __shfl_xor(s, o, 32);
  const float mean = s * (1.f / 64.f);
  const float d0 = m0 - mean, d1 = m1 - mean;
  float vv = d0 * d0 + d1 * d1;
#pragma unroll
  for (int o = 16; o >= 1; o >>= 1) vv += __shfl_xor(vv, o, 32);
  const float inv = rsqrtf(vv * (1.f / 64.f) + 1e-5f);
  const float r0 = fmaxf(d0 * inv * fg[n0] + fb[n0], 0.f);
  const float r1 = fmaxf(d1 * inv * fg[n1] + fb[n1], 0.f);
  __hip_atomic_fetch_add(&agg[ri + n0], r0, __ATOMIC_RELAXED, __HIP_MEMORY_SCOPE_AGENT);
  __hip_atomic_fetch_add(&agg[ri + n1], r1, __ATOMIC_RELAXED, __HIP_MEMORY_SCOPE_AGENT);
}

// ---------------------------------------------------------------------------
// Row LayerNorm over 64 feats (wave per row); optional +deg[row]*dbias (hoisted
// per-edge bias of the Wf stage). Output fp16 node state.
// ---------------------------------------------------------------------------
__global__ __launch_bounds__(256) void ln_rows(
    const float* __restrict__ X, const int* __restrict__ deg,
    const float* __restrict__ dbias, const float* __restrict__ g,
    const float* __restrict__ b, _Float16* __restrict__ Yh, int M)
{
  const int row = (int)blockIdx.x * 8 + (int)(threadIdx.x >> 5);
  if (row >= M) return;
  const int lane = threadIdx.x & 31;
  const int n0 = lane, n1 = lane + 32;
  const size_t r = (size_t)row * 64;
  float x0 = X[r + n0], x1 = X[r + n1];
  if (deg) { const float d = (float)deg[row]; x0 += d * dbias[n0]; x1 += d * dbias[n1]; }
  float s = x0 + x1;
#pragma unroll
  for (int o = 16; o >= 1; o >>= 1) s += __shfl_xor(s, o, 32);
  const float mean = s * (1.f / 64.f);
  const float d0 = x0 - mean, d1 = x1 - mean;
  float vv = d0 * d0 + d1 * d1;
#pragma unroll
  for (int o = 16; o >= 1; o >>= 1) vv += __shfl_xor(vv, o, 32);
  const float inv = rsqrtf(vv * (1.f / 64.f) + 1e-5f);
  Yh[r + n0] = (_Float16)(d0 * inv * g[n0] + b[n0]);
  Yh[r + n1] = (_Float16)(d1 * inv * g[n1] + b[n1]);
}

__global__ void cvt_f32_f16(const float* __restrict__ X, _Float16* __restrict__ Y, int n4)
{
  int i = (int)blockIdx.x * blockDim.x + threadIdx.x;
  const int stride = (int)gridDim.x * blockDim.x;
  for (; i < n4; i += stride) {
    const v4f x = *(const v4f*)(X + (size_t)i * 4);
    v4h o; o[0] = (_Float16)x[0]; o[1] = (_Float16)x[1];
    o[2] = (_Float16)x[2]; o[3] = (_Float16)x[3];
    *(v4h*)(Y + (size_t)i * 4) = o;
  }
}

__global__ void deg_count(const int* __restrict__ idx, int* __restrict__ deg, int E)
{
  const int e = (int)blockIdx.x * blockDim.x + threadIdx.x;
  if (e < E) atomicAdd(&deg[idx[e]], 1);
}

// Input LN (tiny feature dims) fused with first embedding layer (KIN->64, relu).
template <int KIN>
__global__ __launch_bounds__(256) void emb_in(
    const float* __restrict__ X, const float* __restrict__ gin, const float* __restrict__ bin,
    const float* __restrict__ W1, const float* __restrict__ b1,
    _Float16* __restrict__ H, int N)
{
  const int idx = (int)blockIdx.x * 256 + threadIdx.x;
  const int node = idx >> 6, o = idx & 63;
  if (node >= N) return;
  float xs[KIN];
  float mean = 0.f;
#pragma unroll
  for (int k = 0; k < KIN; ++k) { xs[k] = X[(size_t)node * KIN + k]; mean += xs[k]; }
  mean *= (1.f / KIN);
  float var = 0.f;
#pragma unroll
  for (int k = 0; k < KIN; ++k) { const float d = xs[k] - mean; var += d * d; }
  const float inv = rsqrtf(var * (1.f / KIN) + 1e-5f);
  float acc = b1[o];
#pragma unroll
  for (int k = 0; k < KIN; ++k)
    acc += ((xs[k] - mean) * inv * gin[k] + bin[k]) * W1[k * 64 + o];
  H[(size_t)node * 64 + o] = (_Float16)fmaxf(acc, 0.f);
}

__global__ void out_head(const _Float16* __restrict__ H, const float* __restrict__ W2,
                         float* __restrict__ out, int N)
{
  const int n = (int)blockIdx.x * blockDim.x + threadIdx.x;
  if (n >= N) return;
  const _Float16* h = H + (size_t)n * 64;
  float acc = 0.f;
#pragma unroll
  for (int c = 0; c < 64; c += 8) {
    const v8h hv = *(const v8h*)(h + c);
#pragma unroll
    for (int t = 0; t < 8; ++t) acc += (float)hv[t] * W2[c + t];
  }
  out[n] = acc;
}

// ---------------------------------------------------------------------------
struct ConvP {
  const float *We, *Wf, *Wl, *Wo1, *Wo2, *Wr, *bf, *bl, *bo1, *bo2,
              *fin_b, *fin_g, *post_b, *post_g;
};

extern "C" void kernel_launch(void* const* d_in, const int* in_sizes, int n_in,
                              void* d_out, int out_size, void* d_ws, size_t ws_size,
                              hipStream_t stream)
{
  (void)n_in; (void)out_size; (void)ws_size;
  const float* cons_feat = (const float*)d_in[0];
  const int*   ei        = (const int*)d_in[1];
  const float* var_feat  = (const float*)d_in[3];
  const int NC = in_sizes[0] / 5;
  const int E  = in_sizes[2];
  const int NV = in_sizes[3] / 19;
  const int* cons_idx = ei;       // edge_indices[0]
  const int* var_idx  = ei + E;   // edge_indices[1]

  auto F = [&](int i) -> const float* { return (const float*)d_in[i]; };
  // Param leaf order (jax pytree: sorted keys), offset by 4 data inputs:
  // 4..7 cons_emb{W1,W2,b1,b2}; 8,9 cons_norm_in{b,g}; 10..13 cons_norms[0..1]{b,g};
  // 14..27/28..41 conv_c_to_v[0/1]; 42..55/56..69 conv_v_to_c[0/1];
  // 70,71 edge_norm{b,g}; 72..74 out{W1,W2,b1}; 75..78 var_emb{W1,W2,b1,b2};
  // 79,80 var_norm_in{b,g}; 81..84 var_norms[0..1]{b,g}.
  auto conv_at = [&](int k) {
    ConvP p;
    p.We = F(k);      p.Wf = F(k + 1);  p.Wl = F(k + 2);  p.Wo1 = F(k + 3);
    p.Wo2 = F(k + 4); p.Wr = F(k + 5);  p.bf = F(k + 6);  p.bl = F(k + 7);
    p.bo1 = F(k + 8); p.bo2 = F(k + 9); p.fin_b = F(k + 10); p.fin_g = F(k + 11);
    p.post_b = F(k + 12); p.post_g = F(k + 13);
    return p;
  };
  const float* eb = F(70);  // edge LN of 1-elem axis == bias exactly

  // Workspace carve
  size_t off = 0;
  auto carve = [&](size_t bytes) -> char* {
    char* p = (char*)d_ws + off;
    off += (bytes + 255) & ~(size_t)255;
    return p;
  };
  float*     PR   = (float*)carve((size_t)NV * 64 * 4);
  float*     PL   = (float*)carve((size_t)NV * 64 * 4);
  float*     AGG  = (float*)carve((size_t)NV * 64 * 4);
  _Float16*  Ch   = (_Float16*)carve((size_t)NC * 64 * 2);
  _Float16*  Vh   = (_Float16*)carve((size_t)NV * 64 * 2);
  _Float16*  XA   = (_Float16*)carve((size_t)NV * 64 * 2);
  _Float16*  XB   = (_Float16*)carve((size_t)NV * 64 * 2);
  _Float16*  HH   = (_Float16*)carve((size_t)NV * 64 * 2);
  int*       DEGC = (int*)carve((size_t)NC * 4);
  int*       DEGV = (int*)carve((size_t)NV * 4);

  // GEMM variants (compile-time epilogues):
  auto gemm_f      = [&](const _Float16* A, const float* W, float* Yf, int M) {
    gemm64<0,0,0,1,0><<<(M + 127) / 128, 256, 0, stream>>>(A, W, nullptr, nullptr, Yf, nullptr, M);
  };
  auto gemm_fb     = [&](const _Float16* A, const float* W, const float* bias, float* Yf, int M) {
    gemm64<0,1,0,1,0><<<(M + 127) / 128, 256, 0, stream>>>(A, W, bias, nullptr, Yf, nullptr, M);
  };
  auto gemm_h_br   = [&](const _Float16* A, const float* W, const float* bias, _Float16* Yh, int M) {
    gemm64<0,1,1,0,1><<<(M + 127) / 128, 256, 0, stream>>>(A, W, bias, nullptr, nullptr, Yh, M);
  };
  auto gemm_h_cbr  = [&](const _Float16* A, const float* W, const float* bias,
                         const float* Cin, _Float16* Yh, int M) {
    gemm64<1,1,1,0,1><<<(M + 127) / 128, 256, 0, stream>>>(A, W, bias, Cin, nullptr, Yh, M);
  };

  // ---- Input embeddings ----
  emb_in<5><<<(NC * 64 + 255) / 256, 256, 0, stream>>>(cons_feat, F(9), F(8), F(4), F(6), XA, NC);
  gemm_h_br(XA, F(5), F(7), Ch, NC);
  emb_in<19><<<(NV * 64 + 255) / 256, 256, 0, stream>>>(var_feat, F(80), F(79), F(75), F(77), XA, NV);
  gemm_h_br(XA, F(76), F(78), Vh, NV);

  // ---- Degrees (constant across layers; needed for the hoisted Wf bias) ----
  hipMemsetAsync(DEGC, 0, (size_t)NC * 4, stream);
  hipMemsetAsync(DEGV, 0, (size_t)NV * 4, stream);
  deg_count<<<(E + 255) / 256, 256, 0, stream>>>(cons_idx, DEGC, E);
  deg_count<<<(E + 255) / 256, 256, 0, stream>>>(var_idx, DEGV, E);

  // ---- One bipartite conv (updates the "right" node state in place) ----
  auto run_conv = [&](const _Float16* Lh, _Float16* Rh, int Nl, int Nr,
                      const int* tgt, const int* src, const int* degR,
                      const ConvP& p, const float* ng, const float* nb) {
    gemm_fb(Rh, p.Wl, p.bl, PR, Nr);                              // Pr = R@Wl + bl
    gemm_f (Lh, p.Wr, PL, Nl);                                    // Pl = L@Wr
    hipMemsetAsync(AGG, 0, (size_t)Nr * 64 * 4, stream);
    edge_msg<<<(E + 7) / 8, 256, 0, stream>>>(tgt, src, PR, PL, p.We, eb,
                                              p.fin_g, p.fin_b, AGG, E);
    cvt_f32_f16<<<2048, 256, 0, stream>>>(AGG, XA, Nr * 16);
    gemm_f (XA, p.Wf, PL, Nr);                                    // t1 = agg@Wf
    ln_rows<<<(Nr + 7) / 8, 256, 0, stream>>>(PL, degR, p.bf,     // +deg*bf, LN(post)
                                              p.post_g, p.post_b, XB, Nr);
    gemm_f (XB, p.Wo1, PR, Nr);                                   // concat-half 1
    gemm_h_cbr(Rh, p.Wo1 + 64 * 64, p.bo1, PR, HH, Nr);           // + half 2, relu
    gemm_fb(HH, p.Wo2, p.bo2, AGG, Nr);                           // t3 = h@Wo2+bo2
    ln_rows<<<(Nr + 7) / 8, 256, 0, stream>>>(AGG, nullptr, nullptr, ng, nb, Rh, Nr);
  };

  for (int l = 0; l < 2; ++l) {
    // v->c: target = cons (edge_indices row 0), source = var
    run_conv(Vh, Ch, NV, NC, cons_idx, var_idx, DEGC, conv_at(42 + l * 14),
             F(11 + l * 2), F(10 + l * 2));
    // c->v: target = var (edge_indices row 1), source = cons
    run_conv(Ch, Vh, NC, NV, var_idx, cons_idx, DEGV, conv_at(14 + l * 14),
             F(82 + l * 2), F(81 + l * 2));
  }

  // ---- Output head: relu(v@W1+b1) @ W2 ----
  gemm_h_br(Vh, F(72), F(74), XA, NV);
  out_head<<<(NV + 255) / 256, 256, 0, stream>>>(XA, F(73), (float*)d_out, NV);
}
